// SAM_15187004358990
// MI455X (gfx1250) — matrix operand order repaired and checked by reading, not compiled, not run on hardware
//
#include <hip/hip_runtime.h>
#include <hip/hip_bf16.h>
#include <math.h>

typedef __attribute__((ext_vector_type(16))) _Float16 v16h;
typedef __attribute__((ext_vector_type(8)))  float    v8f;
typedef __attribute__((ext_vector_type(4)))  int      v4i;

union AFrag { v16h h; v4i q[2]; };

// ---------------------------------------------------------------------------
// Weight packing: OIHW fp32 -> WMMA B-fragment-major f16.
// GEMM: D[pixel][cout] += A[pixel][k] * B[k][cout],  k = (kh*3+kw)*CIN + ci.
// B fragment (v_wmma_f32_16x16x32_f16, 32x16 B): element (lane L, half j)
//   = B[kc*32 + (j>>3)*16 + (L>>4)*8 + (j&7)][n],  n = nt*16 + (L&15).
// Packed layout: pw[(((kc*2+nt)*32 + L)*16) + j]  -> one 32B load per lane.
// ---------------------------------------------------------------------------
__global__ void pack_weights(const float* __restrict__ w, _Float16* __restrict__ pw,
                             int CIN, int total) {
    int d = blockIdx.x * blockDim.x + threadIdx.x;
    if (d >= total) return;
    int j  = d & 15;
    int L  = (d >> 4) & 31;
    int nt = (d >> 9) & 1;
    int kc = d >> 10;
    int k  = kc * 32 + ((j >> 3) << 4) + ((L >> 4) << 3) + (j & 7);
    int n  = (nt << 4) + (L & 15);
    int tap = k / CIN;          // kh*3+kw
    int ci  = k - tap * CIN;
    pw[d] = (_Float16)w[(n * CIN + ci) * 9 + tap];
}

// ---------------------------------------------------------------------------
// Implicit-GEMM 3x3 SAME conv, Cout=32, via v_wmma_f32_16x16x32_f16.
// Block = 128 threads (4 waves). Block tile = 16 rows x 16 cols of pixels.
// Wave wv computes rows y0+wv*4 .. y0+wv*4+3 : 4 M-tiles (16 px each) x
// 2 N-tiles (32 channels) = 8 accumulators. Each B fragment is reused by
// 4 WMMAs -> 4x less L2 weight traffic than a 1-M-tile wave.
// LDS input tile: [18 rows][18 cols][CIN] f16 (halo included, zero padded).
// A fragment per lane: M = L&15 (pixel), khalf = L>>4:
//   halves j=0..7  -> K = khalf*8 + j        (ds_load_b128 #1)
//   halves j=8..15 -> K = 16 + khalf*8 + j-8 (ds_load_b128 #2)
// D (16x16 f32): element (vgpr v, lane L): M = v + 8*(L>>4), N = L&15.
// Output transposed through LDS for coalesced planar NCHW stores.
// ---------------------------------------------------------------------------
template <int CIN, bool INF16, bool RELU>
__global__ __launch_bounds__(128) void conv3x3_wmma(const void* __restrict__ inp,
                                                    const _Float16* __restrict__ pw,
                                                    float* __restrict__ outp) {
    constexpr int H = 256, W = 256;
    constexpr int KC  = 9 * CIN / 32;   // 32-wide K chunks
    constexpr int TPR = CIN / 32;       // chunks per 3x3 tap

    __shared__ _Float16 sIn[18 * 18 * CIN];
    __shared__ float    sOut[4 * 32 * 17];

    const int tid   = threadIdx.x;
    const int bx    = blockIdx.x;
    const int tileX = bx & 15;
    const int tileY = (bx >> 4) & 15;
    const int b     = bx >> 8;
    const int x0    = tileX * 16;
    const int y0    = tileY * 16;

    // ---- stage input tile (with halo) into LDS as f16, [r][c][ci] ----
    constexpr int TOT = 18 * 18 * CIN;
    for (int e = tid; e < TOT; e += 128) {
        int ci  = e / 324;              // 324 = 18*18
        int rem = e - ci * 324;
        int r   = rem / 18;
        int c   = rem - r * 18;
        int y   = y0 + r - 1;
        int x   = x0 + c - 1;
        _Float16 v = (_Float16)0.0f;
        if ((unsigned)y < (unsigned)H && (unsigned)x < (unsigned)W) {
            size_t gi = ((size_t)(b * CIN + ci) * H + y) * W + x;
            if (INF16) v = ((const _Float16*)inp)[gi];
            else       v = (_Float16)((const float*)inp)[gi];
        }
        sIn[(r * 18 + c) * CIN + ci] = v;
    }
    __syncthreads();

    const int wv  = tid >> 5;
    const int L   = tid & 31;
    const int m   = L & 15;    // A: pixel col of fragment; D: channel-within-halftile
    const int kh2 = L >> 4;

    v8f acc[4][2] = {};

#pragma unroll
    for (int kc = 0; kc < KC; ++kc) {
        const int tap = kc / TPR;
        const int cib = kc - tap * TPR;
        const int kh  = tap / 3;
        const int kw  = tap - kh * 3;
        const int c   = m + kw;

        AFrag A[4];
#pragma unroll
        for (int mt = 0; mt < 4; ++mt) {
            const int r    = wv * 4 + mt + kh;
            const int base = (r * 18 + c) * CIN + cib * 32;   // half units, 16B aligned
            A[mt].q[0] = *(const v4i*)(sIn + base + kh2 * 8);
            A[mt].q[1] = *(const v4i*)(sIn + base + 16 + kh2 * 8);
        }

        v16h B0 = *(const v16h*)(pw + (size_t)(((kc * 2 + 0) * 32 + L) * 16));
        v16h B1 = *(const v16h*)(pw + (size_t)(((kc * 2 + 1) * 32 + L) * 16));

#pragma unroll
        for (int mt = 0; mt < 4; ++mt) {
            acc[mt][0] = __builtin_amdgcn_wmma_f32_16x16x32_f16(false, A[mt].h, false, B0,
                                                                (short)0, acc[mt][0], false, false);
            acc[mt][1] = __builtin_amdgcn_wmma_f32_16x16x32_f16(false, A[mt].h, false, B1,
                                                                (short)0, acc[mt][1], false, false);
        }
    }

    // ---- transpose D through LDS, then coalesced planar stores (4 passes) ----
    float* so = sOut + wv * 32 * 17;
    const int pxi = L & 15;
    const int hlf = L >> 4;
#pragma unroll
    for (int mt = 0; mt < 4; ++mt) {
        __syncthreads();                       // sOut free from previous pass
#pragma unroll
        for (int vi = 0; vi < 8; ++vi) {
            int px = vi + (kh2 << 3);
            float a0 = acc[mt][0][vi], a1 = acc[mt][1][vi];
            if (RELU) { a0 = fmaxf(a0, 0.0f); a1 = fmaxf(a1, 0.0f); }
            so[m * 17 + px]        = a0;       // channels 0..15
            so[(16 + m) * 17 + px] = a1;       // channels 16..31
        }
        __syncthreads();
        const int y = y0 + wv * 4 + mt;
#pragma unroll
        for (int i = 0; i < 16; ++i) {
            int ch = i * 2 + hlf;
            outp[((size_t)(b * 32 + ch) * H + y) * W + x0 + pxi] = so[ch * 17 + pxi];
        }
    }
}

// ---------------------------------------------------------------------------
// Vertical IRNN scan (axis=H). One block per (b, c): 256 threads = 256 columns.
// out[first] = x[first]; out[t] = relu(wc*out[prev] + bc + x[t]). f16 output.
// Fully coalesced (lane index == w).
// ---------------------------------------------------------------------------
__global__ __launch_bounds__(256) void scan_vert(const float* __restrict__ xin,
                                                 _Float16* __restrict__ outp,
                                                 const float* __restrict__ wvp,
                                                 const float* __restrict__ bvp,
                                                 int slot, int reverse) {
    const int b = blockIdx.x >> 5;
    const int c = blockIdx.x & 31;
    const int w = threadIdx.x;
    const float wc = wvp[c];
    const float bc = bvp[c];
    const size_t ip = ((size_t)(b * 32 + c) << 16) + w;
    const size_t op = ((size_t)(b * 128 + slot * 32 + c) << 16) + w;
    if (!reverse) {
        float h = xin[ip];
        outp[op] = (_Float16)h;
        for (int t = 1; t < 256; ++t) {
            h = fmaxf(wc * h + bc + xin[ip + (size_t)t * 256], 0.0f);
            outp[op + (size_t)t * 256] = (_Float16)h;
        }
    } else {
        float h = xin[ip + (size_t)255 * 256];
        outp[op + (size_t)255 * 256] = (_Float16)h;
        for (int t = 254; t >= 0; --t) {
            h = fmaxf(wc * h + bc + xin[ip + (size_t)t * 256], 0.0f);
            outp[op + (size_t)t * 256] = (_Float16)h;
        }
    }
}

// ---------------------------------------------------------------------------
// Horizontal IRNN scan (axis=W). One block per (b, c) plane; 256 threads = 256
// rows. W processed in 32-column chunks staged through LDS (stride 33 padding
// -> conflict-free) so global loads/stores stay coalesced; each thread walks
// its own row inside LDS. Carry kept in a register across chunks.
// ---------------------------------------------------------------------------
__global__ __launch_bounds__(256) void scan_horiz(const float* __restrict__ xin,
                                                  _Float16* __restrict__ outp,
                                                  const float* __restrict__ wvp,
                                                  const float* __restrict__ bvp,
                                                  int slot, int reverse) {
    __shared__ float    ti[256 * 33];
    __shared__ _Float16 to[256 * 33];
    const int b   = blockIdx.x >> 5;
    const int c   = blockIdx.x & 31;
    const int tid = threadIdx.x;
    const float wc = wvp[c];
    const float bc = bvp[c];
    const size_t ipl = ((size_t)(b * 32 + c) << 16);
    const size_t opl = ((size_t)(b * 128 + slot * 32 + c) << 16);

    float h = 0.0f;
    for (int chk = 0; chk < 8; ++chk) {
        const int w0 = reverse ? (224 - chk * 32) : (chk * 32);
        __syncthreads();                 // 'to' of prev chunk fully stored
        for (int i = 0; i < 32; ++i) {   // coalesced 128B row-segment loads
            int flat = i * 256 + tid;
            int r = flat >> 5, cc = flat & 31;
            ti[r * 33 + cc] = xin[ipl + (size_t)r * 256 + w0 + cc];
        }
        __syncthreads();
        for (int j = 0; j < 32; ++j) {   // sequential recurrence in LDS
            int cc = reverse ? (31 - j) : j;
            int w  = w0 + cc;
            float xv = ti[tid * 33 + cc];
            bool first = reverse ? (w == 255) : (w == 0);
            h = first ? xv : fmaxf(wc * h + bc + xv, 0.0f);
            to[tid * 33 + cc] = (_Float16)h;
        }
        __syncthreads();
        for (int i = 0; i < 32; ++i) {   // coalesced f16 stores
            int flat = i * 256 + tid;
            int r = flat >> 5, cc = flat & 31;
            outp[opl + (size_t)r * 256 + w0 + cc] = to[r * 33 + cc];
        }
    }
}

// ---------------------------------------------------------------------------
// 1x1 conv (32 -> 1) + sigmoid.
// ---------------------------------------------------------------------------
__global__ __launch_bounds__(256) void conv_out_sigmoid(const float* __restrict__ inp,
                                                        const float* __restrict__ wo,
                                                        float* __restrict__ outp) {
    int p = blockIdx.x * 256 + threadIdx.x;       // 0 .. 8*65536-1
    int b   = p >> 16;
    int pix = p & 65535;
    float acc = 0.0f;
#pragma unroll
    for (int c = 0; c < 32; ++c)
        acc += inp[((size_t)(b * 32 + c) << 16) + pix] * wo[c];
    outp[p] = 1.0f / (1.0f + expf(-acc));
}

// ---------------------------------------------------------------------------
// Pipeline driver. Workspace layout (needs >= ~192.2 MB):
//   [0,   64MB)  : fp32 conv output buffer (reused: out1 -> out2 -> out3)
//   [64, 192MB)  : f16 IRNN output buffer, 128 channels (reused: r1 -> r2)
//   [192MB, ...) : packed f16 weights (18KB + 72KB + 72KB)
// ---------------------------------------------------------------------------
extern "C" void kernel_launch(void* const* d_in, const int* in_sizes, int n_in,
                              void* d_out, int out_size, void* d_ws, size_t ws_size,
                              hipStream_t stream) {
    const float* x     = (const float*)d_in[0];
    const float* w_in  = (const float*)d_in[1];
    const float* w2    = (const float*)d_in[2];
    const float* w3    = (const float*)d_in[3];
    const float* w_out = (const float*)d_in[4];
    const float* r1w   = (const float*)d_in[5];
    const float* r1b   = (const float*)d_in[6];
    const float* r2w   = (const float*)d_in[7];
    const float* r2b   = (const float*)d_in[8];

    char* ws = (char*)d_ws;
    float*    bufConv = (float*)ws;                               // 64 MB
    _Float16* bufRnn  = (_Float16*)(ws + ((size_t)64 << 20));     // 128 MB
    _Float16* pw1     = (_Float16*)(ws + ((size_t)192 << 20));    // 9*32*32   = 9216 halfs
    _Float16* pw2     = pw1 + 9216;                               // 9*128*32  = 36864 halfs
    _Float16* pw3     = pw2 + 36864;

    // pack weights into WMMA B-fragment order
    pack_weights<<<(9216 + 255) / 256, 256, 0, stream>>>(w_in, pw1, 32, 9216);
    pack_weights<<<(36864 + 255) / 256, 256, 0, stream>>>(w2, pw2, 128, 36864);
    pack_weights<<<(36864 + 255) / 256, 256, 0, stream>>>(w3, pw3, 128, 36864);

    // conv_in: 32 -> 32 (fp32 input, converted to f16 while staging LDS)
    conv3x3_wmma<32, false, false><<<2048, 128, 0, stream>>>((const void*)x, pw1, bufConv);

    // irnn1: up(rev,V,0), right(fwd,H,1), down(fwd,V,2), left(rev,H,3)
    scan_vert <<<256, 256, 0, stream>>>(bufConv, bufRnn, r1w + 0 * 32, r1b + 0 * 32, 0, 1);
    scan_horiz<<<256, 256, 0, stream>>>(bufConv, bufRnn, r1w + 1 * 32, r1b + 1 * 32, 1, 0);
    scan_vert <<<256, 256, 0, stream>>>(bufConv, bufRnn, r1w + 2 * 32, r1b + 2 * 32, 2, 0);
    scan_horiz<<<256, 256, 0, stream>>>(bufConv, bufRnn, r1w + 3 * 32, r1b + 3 * 32, 3, 1);

    // conv2: 128 -> 32 (f16 input)
    conv3x3_wmma<128, true, false><<<2048, 128, 0, stream>>>((const void*)bufRnn, pw2, bufConv);

    // irnn2
    scan_vert <<<256, 256, 0, stream>>>(bufConv, bufRnn, r2w + 0 * 32, r2b + 0 * 32, 0, 1);
    scan_horiz<<<256, 256, 0, stream>>>(bufConv, bufRnn, r2w + 1 * 32, r2b + 1 * 32, 1, 0);
    scan_vert <<<256, 256, 0, stream>>>(bufConv, bufRnn, r2w + 2 * 32, r2b + 2 * 32, 2, 0);
    scan_horiz<<<256, 256, 0, stream>>>(bufConv, bufRnn, r2w + 3 * 32, r2b + 3 * 32, 3, 1);

    // conv3: 128 -> 32 + ReLU
    conv3x3_wmma<128, true, true><<<2048, 128, 0, stream>>>((const void*)bufRnn, pw3, bufConv);

    // 1x1 conv + sigmoid -> mask (8,1,256,256) fp32
    conv_out_sigmoid<<<2048, 256, 0, stream>>>(bufConv, w_out, (float*)d_out);
}
